// Seq2SeqAutoencoder_78460462563469
// MI455X (gfx1250) — matrix operand order repaired and checked by reading, not compiled, not run on hardware
//
#include <hip/hip_runtime.h>

typedef __attribute__((ext_vector_type(16))) _Float16 v16h;
typedef __attribute__((ext_vector_type(8)))  _Float16 v8h;
typedef __attribute__((ext_vector_type(4)))  _Float16 v4h;
typedef __attribute__((ext_vector_type(8)))  float    v8f;
typedef __attribute__((ext_vector_type(4)))  float    v4f;

// Problem constants (from reference)
static constexpr int HD     = 64;          // hidden dim
static constexpr int DIN    = 32;          // input dim
static constexpr int KTOT   = DIN + HD;    // 96 : concat [x, h] K-dim
static constexpr int NG     = 4 * HD;      // 256: gate columns (i,f,g,o)
static constexpr int TSTEPS = 500;
static constexpr int BATCH  = 4096;

// Tiling
static constexpr int WAVES          = 2;                    // waves per block
static constexpr int ROWS_PER_WAVE  = 16;                   // M-tile
static constexpr int ROWS_PER_BLOCK = WAVES * ROWS_PER_WAVE;

// ---- nonlinearities: prefer native V_TANH_F32 (CDNA5 transcendental) ----
__device__ __forceinline__ float fast_tanh(float x) {
#if __has_builtin(__builtin_amdgcn_tanhf)
    return __builtin_amdgcn_tanhf(x);
#elif __has_builtin(__builtin_amdgcn_tanh_f32)
    return __builtin_amdgcn_tanh_f32(x);
#else
    float e = __expf(2.0f * x);
  #if __has_builtin(__builtin_amdgcn_rcpf)
    return 1.0f - 2.0f * __builtin_amdgcn_rcpf(e + 1.0f);
  #else
    return (e - 1.0f) / (e + 1.0f);
  #endif
#endif
}
__device__ __forceinline__ float fast_sigmoid(float x) {
    return 0.5f + 0.5f * fast_tanh(0.5f * x);   // sigmoid via tanh: 1 trans op + fma
}

// A fragment: 16x32 f16, from row-major [16][KTOT] LDS buffer.
// Layout: lanes 0-15 row M=lane hold K {0..15}; lanes 16-31 hold K {16..31} (matches B fragment).
__device__ __forceinline__ v16h load_afrag(const _Float16* base, int kbase, int l, int hi) {
    const _Float16* p = base + l * KTOT + kbase + hi * 16;
    v8h lo = *(const v8h*)(p);
    v8h hh = *(const v8h*)(p + 8);
    v16h a;
#pragma unroll
    for (int i = 0; i < 8; ++i) { a[i] = lo[i]; a[i + 8] = hh[i]; }
    return a;
}

// B fragment: 32x16 f16 of B[k][n] = W[n][k]; W stored row-major [N][kstride] f16 in LDS.
// Lane (group hi, l): column n = n0+l, 16 contiguous K values starting at kbase + hi*16.
// Deliberately non-volatile: MachineLICM hoists all 48 fragments of the LSTM weight
// matrix (384 VGPRs) out of the 500-iteration time loop, keeping weights register-
// resident. That puts the wave at ~760 VGPRs (1 wave/SIMD), which is the intended
// operating point: per-step LDS traffic shrinks to the A-fragments + h feedback, and
// WMMA (tracked as TRANS) co-executes with the gate VALU within the single wave.
__device__ __forceinline__ v16h load_bfrag(const _Float16* w, int kstride, int n0,
                                           int kbase, int l, int hi) {
    const _Float16* p = w + (size_t)(n0 + l) * kstride + kbase + hi * 16;
    return *(const v16h*)(p);
}

// One LSTM step for a 16-row batch tile.
// inb: row-major [16][KTOT] f16; cols [0,DIN) = x_t, cols [DIN,KTOT) = h_{t-1}.
// Writes h_t (f16) back into cols [DIN,KTOT) of inb; updates c in registers.
__device__ __forceinline__ void lstm_step(const _Float16* __restrict__ inb_r,
                                          _Float16* __restrict__ inb_w,
                                          const _Float16* __restrict__ w,
                                          const float* __restrict__ bias,
                                          v8f (&c)[4], int l, int hi) {
    v8f acc[16];
#pragma unroll
    for (int nt = 0; nt < 16; ++nt) {
        float bv = bias[nt * 16 + l];
#pragma unroll
        for (int i = 0; i < 8; ++i) acc[nt][i] = bv;
    }
#pragma unroll
    for (int kt = 0; kt < 3; ++kt) {
        v16h a = load_afrag(inb_r, kt * 32, l, hi);
#pragma unroll
        for (int nt = 0; nt < 16; ++nt) {
            v16h b = load_bfrag(w, KTOT, nt * 16, kt * 32, l, hi);
            acc[nt] = __builtin_amdgcn_wmma_f32_16x16x32_f16(
                false, a, false, b, (short)0, acc[nt], false, false);
        }
    }
    // Gate order i,f,g,o along N: tiles 0-3 = i, 4-7 = f, 8-11 = g, 12-15 = o.
#pragma unroll
    for (int j = 0; j < 4; ++j) {
#pragma unroll
        for (int r = 0; r < 8; ++r) {
            float iv = fast_sigmoid(acc[j][r]);
            float fv = fast_sigmoid(acc[4 + j][r]);
            float gv = fast_tanh(acc[8 + j][r]);
            float ov = fast_sigmoid(acc[12 + j][r]);
            float cn = fv * c[j][r] + iv * gv;
            c[j][r] = cn;
            float hv = ov * fast_tanh(cn);
            int M = r + hi * 8;
            inb_w[M * KTOT + DIN + j * 16 + l] = (_Float16)hv;
        }
    }
}

__device__ __forceinline__ void stage_lstm_weights(const float* __restrict__ Wih,
                                                   const float* __restrict__ Whh,
                                                   const float* __restrict__ b,
                                                   _Float16* __restrict__ sW,
                                                   float* __restrict__ sB,
                                                   int tid, int nthreads) {
    for (int idx = tid; idx < NG * KTOT; idx += nthreads) {
        int n = idx / KTOT, k = idx - n * KTOT;
        float v = (k < DIN) ? Wih[n * DIN + k] : Whh[n * HD + (k - DIN)];
        sW[idx] = (_Float16)v;
    }
    for (int n = tid; n < NG; n += nthreads) sB[n] = b[n];
}

__global__ __launch_bounds__(WAVES * 32)
void seq2seq_lstm_kernel(const float* __restrict__ x,
                         const float* __restrict__ Wih_e, const float* __restrict__ Whh_e,
                         const float* __restrict__ b_e,
                         const float* __restrict__ Wih_d, const float* __restrict__ Whh_d,
                         const float* __restrict__ b_d,
                         const float* __restrict__ Wout, const float* __restrict__ bout,
                         float* __restrict__ out) {
    __shared__ __align__(32) _Float16 sW[NG * KTOT];              // 48 KB (reused enc -> dec)
    __shared__ __align__(32) _Float16 sWout[DIN * HD];            // 4 KB
    __shared__ float sB[NG];                                      // 1 KB
    __shared__ float sBout[DIN];
    __shared__ __align__(32) _Float16 sIn[WAVES * ROWS_PER_WAVE * KTOT]; // 6 KB

    const int tid  = threadIdx.x;
    const int wave = tid >> 5;
    const int lane = tid & 31;
    const int l    = lane & 15;
    const int hi   = lane >> 4;
    const int rowbase = blockIdx.x * ROWS_PER_BLOCK + wave * ROWS_PER_WAVE;
    _Float16* inb = &sIn[wave * ROWS_PER_WAVE * KTOT];

    // ---- stage encoder weights; zero h0/c0 and input buffer ----
    stage_lstm_weights(Wih_e, Whh_e, b_e, sW, sB, tid, blockDim.x);
    for (int i = lane; i < ROWS_PER_WAVE * KTOT; i += 32) inb[i] = (_Float16)0.0f;
    v8f c[4];
#pragma unroll
    for (int j = 0; j < 4; ++j)
#pragma unroll
        for (int r = 0; r < 8; ++r) c[j][r] = 0.0f;
    __syncthreads();

    // ---- encoder: 500 steps over x (double-buffered x_t loads) ----
    {
        const int xrow  = lane >> 1;   // 0..15
        const int xpart = lane & 1;    // half-row of 16 floats
        const float* xbase = x + (size_t)(rowbase + xrow) * TSTEPS * DIN + xpart * 16;
        _Float16* dst = inb + xrow * KTOT + xpart * 16;

        v4f xreg[4];
#pragma unroll
        for (int ch = 0; ch < 4; ++ch) xreg[ch] = *(const v4f*)(xbase + ch * 4);

        for (int t = 0; t < TSTEPS; ++t) {
            // commit x_t (f32 regs -> f16 LDS)
#pragma unroll
            for (int ch = 0; ch < 4; ++ch) {
                v4h hv;
#pragma unroll
                for (int i = 0; i < 4; ++i) hv[i] = (_Float16)xreg[ch][i];
                *(v4h*)(dst + ch * 4) = hv;
            }
            // prefetch t+2 (global_prefetch_b8), load t+1 into regs; both overlap the step
            if (t + 2 < TSTEPS)
                __builtin_prefetch(xbase + (size_t)(t + 2) * DIN, 0, 0);
            if (t + 1 < TSTEPS) {
                const float* xr = xbase + (size_t)(t + 1) * DIN;
#pragma unroll
                for (int ch = 0; ch < 4; ++ch) xreg[ch] = *(const v4f*)(xr + ch * 4);
            }
            lstm_step(inb, inb, sW, sB, c, l, hi);
        }
    }

    // ---- swap in decoder weights + output projection ----
    __syncthreads();   // all waves done reading encoder sW
    stage_lstm_weights(Wih_d, Whh_d, b_d, sW, sB, tid, blockDim.x);
    for (int idx = tid; idx < DIN * HD; idx += blockDim.x)
        sWout[idx] = (_Float16)Wout[idx];                  // row-major [32][64]
    for (int idx = tid; idx < DIN; idx += blockDim.x) sBout[idx] = bout[idx];
    // dec_in0 = 0: clear x-region of input buffer (h,c carry over from encoder)
    for (int i = lane; i < ROWS_PER_WAVE * DIN; i += 32) {
        int rr = i / DIN, cc = i - rr * DIN;
        inb[rr * KTOT + cc] = (_Float16)0.0f;
    }
    __syncthreads();

    // ---- autoregressive decoder: 500 steps ----
    for (int t = 0; t < TSTEPS; ++t) {
        lstm_step(inb, inb, sW, sB, c, l, hi);

        // out = h @ Wout^T + bout : M=16, N=32 (2 tiles), K=64 (2 tiles, h at col DIN)
        v8f acco[2];
#pragma unroll
        for (int nt = 0; nt < 2; ++nt) {
            float bv = sBout[nt * 16 + l];
#pragma unroll
            for (int i = 0; i < 8; ++i) acco[nt][i] = bv;
        }
#pragma unroll
        for (int kt = 0; kt < 2; ++kt) {
            v16h a = load_afrag(inb, DIN + kt * 32, l, hi);
#pragma unroll
            for (int nt = 0; nt < 2; ++nt) {
                v16h b = load_bfrag(sWout, HD, nt * 16, kt * 32, l, hi);
                acco[nt] = __builtin_amdgcn_wmma_f32_16x16x32_f16(
                    false, a, false, b, (short)0, acco[nt], false, false);
            }
        }
        // store out[B,t,D] (f32) and feed back as next decoder input (f16)
#pragma unroll
        for (int nt = 0; nt < 2; ++nt) {
#pragma unroll
            for (int r = 0; r < 8; ++r) {
                int M = r + hi * 8;
                float v = acco[nt][r];
                out[((size_t)(rowbase + M) * TSTEPS + t) * DIN + nt * 16 + l] = v;
                inb[M * KTOT + nt * 16 + l] = (_Float16)v;
            }
        }
    }
}

extern "C" void kernel_launch(void* const* d_in, const int* in_sizes, int n_in,
                              void* d_out, int out_size, void* d_ws, size_t ws_size,
                              hipStream_t stream) {
    (void)in_sizes; (void)n_in; (void)out_size; (void)d_ws; (void)ws_size;
    const float* x     = (const float*)d_in[0];
    const float* Wih_e = (const float*)d_in[1];
    const float* Whh_e = (const float*)d_in[2];
    const float* b_e   = (const float*)d_in[3];
    const float* Wih_d = (const float*)d_in[4];
    const float* Whh_d = (const float*)d_in[5];
    const float* b_d   = (const float*)d_in[6];
    const float* Wout  = (const float*)d_in[7];
    const float* bout  = (const float*)d_in[8];
    float* out = (float*)d_out;

    dim3 grid(BATCH / ROWS_PER_BLOCK);   // 4096 / 32 = 128 blocks
    dim3 block(WAVES * 32);              // 2 waves (wave32)
    seq2seq_lstm_kernel<<<grid, block, 0, stream>>>(
        x, Wih_e, Whh_e, b_e, Wih_d, Whh_d, b_d, Wout, bout, out);
}
